// Attn_87196426043406
// MI455X (gfx1250) — compile-verified, matrix operand order checked by reading, not run on hardware
//
#include <hip/hip_runtime.h>
#include <stdint.h>

// ---------------------------------------------------------------------------
// Attention forward for B=4,H=16,L=S=2048,D=64 (fp32 in, fp32 out).
// Outputs: attn_output [B,H,L,D] then attn_scores [B,H,L,S] in d_out.
//
// Roofline: scores writeback = 1.07 GB => ~46us floor @ 23.3 TB/s; FLOPs are
// negligible on bf16 WMMA. K/V (67MB) are L2-resident (192MB L2).
//
// CDNA5 data path: K/V tiles staged into LDS by the Tensor Data Mover
// (tensor_load_to_lds, TENSORcnt), double-buffered so tile j+1 DMAs while
// tile j feeds v_wmma_f32_16x16x32_bf16. Two-pass softmax per 16-row Q block
// per wave (wave32): pass A = online row max/sum; pass B = recompute logits,
// write normalized scores, accumulate P·V.
// ---------------------------------------------------------------------------

typedef __attribute__((ext_vector_type(16))) __bf16       v16bf;
typedef __attribute__((ext_vector_type(8)))  float        v8f;
typedef __attribute__((ext_vector_type(4)))  unsigned int v4u;
typedef __attribute__((ext_vector_type(8)))  int          v8i;
typedef __attribute__((ext_vector_type(4)))  int          v4i;

#define WMMA_BF16(A, B, C) \
    __builtin_amdgcn_wmma_f32_16x16x32_bf16(false, (A), false, (B), (short)0, (C), false, false)

constexpr int Bc = 4, Hc = 16, Lc = 2048, Sc = 2048, Dc = 64;
constexpr int THREADS = 256;            // 8 waves of 32
constexpr int WAVES   = 8;
constexpr int ROWS_PER_WG = WAVES * 16; // 128
constexpr int NIT = Sc / 32;            // 64 key-tile iterations

// Issue one TDM DMA: contiguous 1-D tile of n_elems fp32 from global -> LDS.
// D# per CDNA5 ISA ch.8: group0 = {count, lds_addr, global_addr(57b), type=2},
// group1 = {data_size=4B, tensor_dim0=tile_dim0=n, dim0_stride=n}, groups 2/3
// zero (<=2-D tensor). This toolchain exposes the 6-arg builtin form.
__device__ __forceinline__ void tdm_load_tile(uint32_t lds_off,
                                              const float* gptr, int n) {
    const uint64_t ga = (uint64_t)(uintptr_t)gptr;
    v4u g0;
    g0.x = 1u;                                            // count=1, user mode
    g0.y = lds_off;                                       // lds_addr (bytes)
    g0.z = (uint32_t)ga;                                  // global_addr[31:0]
    g0.w = (uint32_t)((ga >> 32) & 0x1FFFFFFu) | (2u << 30); // addr[56:32]|type=2
    v8i g1;
    g1[0] = (int)(2u << 16);                              // data_size = 4 bytes
    g1[1] = (int)(((uint32_t)n & 0xFFFFu) << 16);         // tensor_dim0[15:0]
    g1[2] = (int)((((uint32_t)n >> 16) & 0xFFFFu) | (1u << 16)); // dim0 hi | tensor_dim1=1
    g1[3] = (int)(((uint32_t)n & 0xFFFFu) << 16);         // tile_dim0 = n
    g1[4] = 0;                                            // tile_dim1/2 unused
    g1[5] = n;                                            // tensor_dim0_stride
    g1[6] = 0;
    g1[7] = 0;
    const v4i z4 = {0, 0, 0, 0};
    const v8i z8 = {0, 0, 0, 0, 0, 0, 0, 0};
    __builtin_amdgcn_tensor_load_to_lds(g0, g1, z4, z4, z8, 0);
}

__device__ __forceinline__ uint32_t lds_offset(const void* p) {
    return (uint32_t)(uintptr_t)p;   // flat LDS addr: low 32 bits = LDS offset
}

__global__ __launch_bounds__(THREADS)
void attn_fwd(const float* __restrict__ q,
              const float* __restrict__ k,
              const float* __restrict__ v,
              float* __restrict__ out_o,
              float* __restrict__ out_s)
{
    // Double-buffered TDM destinations + per-wave P relayout tile (40 KB).
    __shared__ float  kbuf[2][32 * Dc];               // 2 x 8 KB
    __shared__ float  vbuf[2][32 * Dc];               // 2 x 8 KB
    __shared__ __bf16 pstage[WAVES][16 * 32];         // 8 KB

    const int tid  = threadIdx.x;
    const int wave = tid >> 5;
    const int lane = tid & 31;
    const int ln   = lane & 15;       // index within 16-lane group
    const int half = lane >> 4;       // 0 or 1
    const int kb   = half * 8;        // ISA 16-bit A/B layout: K base per half

    const int mblk = blockIdx.x & 15;    // L / ROWS_PER_WG = 16 blocks
    const int bh   = blockIdx.x >> 4;    // 0..63
    const int m0   = mblk * ROWS_PER_WG + wave * 16;

    const float* kbh = k + (size_t)bh * Sc * Dc;
    const float* vbh = v + (size_t)bh * Sc * Dc;
    const size_t qbase = (size_t)bh * Lc * Dc;
    const size_t sbase = (size_t)bh * (size_t)Lc * Sc;
    const size_t obase = (size_t)bh * Lc * Dc;

    // ---- Q fragments (A layout), pre-scaled by 1/sqrt(64) -----------------
    v16bf a0, a1;
    {
        const float* qrow = q + qbase + (size_t)(m0 + ln) * Dc;
        const float sc = 0.125f;
#pragma unroll
        for (int e = 0; e < 8; ++e) {
            a0[e]     = (__bf16)(qrow[kb + e]      * sc);
            a0[e + 8] = (__bf16)(qrow[16 + kb + e] * sc);
            a1[e]     = (__bf16)(qrow[32 + kb + e] * sc);
            a1[e + 8] = (__bf16)(qrow[48 + kb + e] * sc);
        }
    }

    // ======================= PASS A: row max / sum =========================
    float pmax[8], psum[8];
#pragma unroll
    for (int g = 0; g < 8; ++g) { pmax[g] = -1e30f; psum[g] = 0.0f; }

    if (wave == 0) tdm_load_tile(lds_offset(&kbuf[0][0]), kbh, 32 * Dc);

    for (int j2 = 0; j2 < NIT; ++j2) {
        __syncthreads();   // all readers done with the buffer we overwrite next
        if (wave == 0) {
            const int jn = (j2 + 1) & (NIT - 1);       // wrap: benign prefetch
            tdm_load_tile(lds_offset(&kbuf[(j2 + 1) & 1][0]),
                          kbh + (size_t)jn * 32 * Dc, 32 * Dc);
            __builtin_amdgcn_s_wait_tensorcnt((short)1);  // current tile done
        }
        __syncthreads();
        const float* ktile = &kbuf[j2 & 1][0];

#pragma unroll
        for (int t = 0; t < 2; ++t) {
            const float* krow = &ktile[(t * 16 + ln) * Dc];
            v16bf b0, b1;
#pragma unroll
            for (int e = 0; e < 8; ++e) {
                b0[e]     = (__bf16)krow[kb + e];
                b0[e + 8] = (__bf16)krow[16 + kb + e];
                b1[e]     = (__bf16)krow[32 + kb + e];
                b1[e + 8] = (__bf16)krow[48 + kb + e];
            }
            v8f c = {};
            c = WMMA_BF16(a0, b0, c);
            c = WMMA_BF16(a1, b1, c);
#pragma unroll
            for (int g = 0; g < 8; ++g) {
                const float x  = c[g];
                const float nm = fmaxf(pmax[g], x);
                psum[g] = psum[g] * __expf(pmax[g] - nm) + __expf(x - nm);
                pmax[g] = nm;
            }
        }
    }

    // cross-lane reduction within each 16-lane group (rows stay in-half)
#pragma unroll
    for (int g = 0; g < 8; ++g) {
#pragma unroll
        for (int off = 1; off < 16; off <<= 1) {
            const float om = __shfl_xor(pmax[g], off);
            const float os = __shfl_xor(psum[g], off);
            const float nm = fmaxf(pmax[g], om);
            psum[g] = psum[g] * __expf(pmax[g] - nm) + os * __expf(om - nm);
            pmax[g] = nm;
        }
        psum[g] = 1.0f / psum[g];   // now inverse row sum
    }

    // ================= PASS B: scores out + P·V accumulate =================
    v8f o[4] = {};
    float* srow = out_s + sbase;

    if (wave == 0) {
        tdm_load_tile(lds_offset(&kbuf[0][0]), kbh, 32 * Dc);
        tdm_load_tile(lds_offset(&vbuf[0][0]), vbh, 32 * Dc);
    }

    for (int j2 = 0; j2 < NIT; ++j2) {
        __syncthreads();
        if (wave == 0) {
            const int jn = (j2 + 1) & (NIT - 1);
            const int nb = (j2 + 1) & 1;
            tdm_load_tile(lds_offset(&kbuf[nb][0]), kbh + (size_t)jn * 32 * Dc, 32 * Dc);
            tdm_load_tile(lds_offset(&vbuf[nb][0]), vbh + (size_t)jn * 32 * Dc, 32 * Dc);
            __builtin_amdgcn_s_wait_tensorcnt((short)2);  // current K+V done
        }
        __syncthreads();
        const float* ktile = &kbuf[j2 & 1][0];
        const float* vtile = &vbuf[j2 & 1][0];

        __bf16* ps = &pstage[wave][0];
#pragma unroll
        for (int t = 0; t < 2; ++t) {
            const float* krow = &ktile[(t * 16 + ln) * Dc];
            v16bf b0, b1;
#pragma unroll
            for (int e = 0; e < 8; ++e) {
                b0[e]     = (__bf16)krow[kb + e];
                b0[e + 8] = (__bf16)krow[16 + kb + e];
                b1[e]     = (__bf16)krow[32 + kb + e];
                b1[e + 8] = (__bf16)krow[48 + kb + e];
            }
            v8f c = {};
            c = WMMA_BF16(a0, b0, c);
            c = WMMA_BF16(a1, b1, c);

            const int col = j2 * 32 + t * 16 + ln;
#pragma unroll
            for (int g = 0; g < 8; ++g) {
                const int   M = g + 8 * half;
                const float p = __expf(c[g] - pmax[g]) * psum[g];
                srow[(size_t)(m0 + M) * Sc + col] = p;     // coalesced per g
                ps[M * 32 + t * 16 + ln] = (__bf16)p;      // relayout via LDS
            }
        }

        // A fragment of P (16 rows x 32 keys) from per-wave LDS stage
        v16bf ap;
        {
            const __bf16* prow = &pstage[wave][ln * 32];
#pragma unroll
            for (int e = 0; e < 8; ++e) {
                ap[e]     = prow[kb + e];
                ap[e + 8] = prow[16 + kb + e];
            }
        }

        // O(16x64) += P(16x32) * V(32x64), 4 chunks of N=16
#pragma unroll
        for (int ch = 0; ch < 4; ++ch) {
            v16bf bv;
#pragma unroll
            for (int e = 0; e < 8; ++e) {
                bv[e]     = (__bf16)vtile[(kb + e) * Dc + ch * 16 + ln];
                bv[e + 8] = (__bf16)vtile[(16 + kb + e) * Dc + ch * 16 + ln];
            }
            o[ch] = WMMA_BF16(ap, bv, o[ch]);
        }
    }

    // ---- write output tile ------------------------------------------------
#pragma unroll
    for (int ch = 0; ch < 4; ++ch) {
#pragma unroll
        for (int g = 0; g < 8; ++g) {
            const int M = g + 8 * half;
            out_o[obase + (size_t)(m0 + M) * Dc + ch * 16 + ln] = o[ch][g];
        }
    }
}

extern "C" void kernel_launch(void* const* d_in, const int* in_sizes, int n_in,
                              void* d_out, int out_size, void* d_ws, size_t ws_size,
                              hipStream_t stream) {
    (void)in_sizes; (void)n_in; (void)d_ws; (void)ws_size; (void)out_size;
    const float* q = (const float*)d_in[0];
    const float* k = (const float*)d_in[1];
    const float* v = (const float*)d_in[2];
    float* out_o = (float*)d_out;
    float* out_s = out_o + (size_t)Bc * Hc * Lc * Dc;   // scores follow output

    dim3 grid((Bc * Hc) * (Lc / ROWS_PER_WG));  // 64 * 16 = 1024 workgroups
    dim3 block(THREADS);
    hipLaunchKernelGGL(attn_fwd, grid, block, 0, stream, q, k, v, out_o, out_s);
}
